// GPT2Transformer_54683523613130
// MI455X (gfx1250) — compile-verified, matrix operand order checked by reading, not run on hardware
//
#include <hip/hip_runtime.h>
#include <hip/hip_bf16.h>

// ---------------------------------------------------------------------------
// CDNA5 (gfx1250) GPT-2 forward: bf16 WMMA GEMMs + flash attention.
// ---------------------------------------------------------------------------

typedef __attribute__((ext_vector_type(16))) __bf16 v16bf;
typedef __attribute__((ext_vector_type(8)))  float  v8f;

union Frag16 {
    v16bf v;
    unsigned int u[8];
    unsigned short h[16];
};

__device__ __forceinline__ unsigned short f2bf(float f) {
    unsigned int u = __float_as_uint(f);
    unsigned int r = u + 0x7FFFu + ((u >> 16) & 1u);   // round-to-nearest-even
    return (unsigned short)(r >> 16);
}
__device__ __forceinline__ unsigned int pack2bf(float a, float b) {
    return (unsigned int)f2bf(a) | ((unsigned int)f2bf(b) << 16);
}
// CDNA5 16-bit A/B fragment K pattern: lanes 0-15 hold K {0..7, 16..23},
// lanes 16-31 hold K {8..15, 24..31}; VGPR j holds K pair (k, k+1).
__device__ __forceinline__ int kbase(int j, int half) {
    return ((j < 4) ? 2 * j : 2 * j + 8) + half * 8;
}
__device__ __forceinline__ float gelu_new(float x) {
    return 0.5f * x * (1.0f + tanhf(0.7978845608028654f * (x + 0.044715f * x * x * x)));
}

// ---------------------------------------------------------------------------
// Tiled GEMM: C[M,N] = A[M,K] @ W[K,N] + bias, with fused epilogue.
// OP: 0 = bias; 1 = bias + residual(extra); 2 = bias + gelu_new;
//     3 = bias + wpe(extra, pos = m % 1024)
// Block: 256 threads (8 wave32), tile 128x128, BK = 32.
// ---------------------------------------------------------------------------
#define BM 128
#define BN 128
#define BK 32

template <int OP>
__global__ __launch_bounds__(256) void gemm_kernel(
    const float* __restrict__ A, const float* __restrict__ W,
    const float* __restrict__ bias, const float* __restrict__ extra,
    float* __restrict__ C, int M, int N, int K)
{
    __shared__ __attribute__((aligned(16))) unsigned short As[BM][BK + 8]; // [m][k]
    __shared__ __attribute__((aligned(16))) unsigned short Ws[BN][BK + 4]; // [n][k] (transposed)

    const int tid  = threadIdx.x;
    const int lane = tid & 31;
    const int wave = tid >> 5;
    const int half = (lane >> 4) & 1;
    const int l16  = lane & 15;
    const int wm   = wave >> 1;   // 0..3  -> 32-row slice
    const int wn   = wave & 1;    // 0..1  -> 64-col slice
    const int m0   = blockIdx.y * BM;
    const int n0   = blockIdx.x * BN;

    const v8f zero8 = {0.f, 0.f, 0.f, 0.f, 0.f, 0.f, 0.f, 0.f};
    v8f acc[2][4];
#pragma unroll
    for (int r = 0; r < 2; ++r)
#pragma unroll
        for (int f = 0; f < 4; ++f) acc[r][f] = zero8;

    for (int k0 = 0; k0 < K; k0 += BK) {
        // ---- stage A tile (128x32 fp32 -> bf16), float4 loads ----
#pragma unroll
        for (int i = 0; i < 4; ++i) {
            int e  = tid + i * 256;            // float4 index, 1024 total
            int r  = e >> 3;                   // row 0..127
            int c4 = (e & 7) << 2;             // col 0,4,...,28
            const float4 f4 = *(const float4*)(A + (size_t)(m0 + r) * K + k0 + c4);
            *(unsigned int*)&As[r][c4 + 0] = pack2bf(f4.x, f4.y);
            *(unsigned int*)&As[r][c4 + 2] = pack2bf(f4.z, f4.w);
        }
        // ---- stage W tile (32x128 fp32 -> bf16, stored [n][k]) ----
#pragma unroll
        for (int i = 0; i < 4; ++i) {
            int e  = tid + i * 256;
            int r  = e >> 5;                   // k row 0..31
            int c4 = (e & 31) << 2;            // n col 0,4,...,124
            const float4 f4 = *(const float4*)(W + (size_t)(k0 + r) * N + n0 + c4);
            Ws[c4 + 0][r] = f2bf(f4.x);
            Ws[c4 + 1][r] = f2bf(f4.y);
            Ws[c4 + 2][r] = f2bf(f4.z);
            Ws[c4 + 3][r] = f2bf(f4.w);
        }
        // prefetch next K-tile into cache hierarchy
        if (k0 + BK < K) {
            __builtin_prefetch(A + (size_t)(m0 + (tid >> 1)) * K + k0 + BK + (tid & 1) * 16);
            __builtin_prefetch(W + (size_t)(k0 + BK + (tid >> 3)) * N + n0 + (tid & 7) * 16);
        }
        __syncthreads();

        // ---- build fragments and multiply ----
        Frag16 afr[2], bfr[4];
#pragma unroll
        for (int r = 0; r < 2; ++r) {
            int m = wm * 32 + r * 16 + l16;
#pragma unroll
            for (int j = 0; j < 8; ++j)
                afr[r].u[j] = *(const unsigned int*)&As[m][kbase(j, half)];
        }
#pragma unroll
        for (int f = 0; f < 4; ++f) {
            int n = wn * 64 + f * 16 + l16;
#pragma unroll
            for (int j = 0; j < 8; ++j)
                bfr[f].u[j] = *(const unsigned int*)&Ws[n][kbase(j, half)];
        }
#pragma unroll
        for (int r = 0; r < 2; ++r)
#pragma unroll
            for (int f = 0; f < 4; ++f)
                acc[r][f] = __builtin_amdgcn_wmma_f32_16x16x32_bf16(
                    false, afr[r].v, false, bfr[f].v, (short)0, acc[r][f], false, false);
        __syncthreads();
    }

    // ---- epilogue: C-frag layout: row = r*16 + half*8 + i, col = f*16 + l16 ----
#pragma unroll
    for (int r = 0; r < 2; ++r) {
#pragma unroll
        for (int f = 0; f < 4; ++f) {
            int n = n0 + wn * 64 + f * 16 + l16;
            float bn = bias[n];
#pragma unroll
            for (int i = 0; i < 8; ++i) {
                int m = m0 + wm * 32 + r * 16 + half * 8 + i;
                float v = acc[r][f][i] + bn;
                if (OP == 1) v += extra[(size_t)m * N + n];
                if (OP == 2) v = gelu_new(v);
                if (OP == 3) v += extra[(size_t)(m & 1023) * N + n];  // wpe[m % S][n]
                C[(size_t)m * N + n] = v;
            }
        }
    }
}

// ---------------------------------------------------------------------------
// Flash attention. Grid: (S/64, H, NSEQ). Block: 128 threads (4 wave32),
// each wave owns 16 query rows. Mask: causal + "odd q drops k = q-1"
// (both only ever apply on the diagonal 64-wide k-tile).
// ---------------------------------------------------------------------------
__global__ __launch_bounds__(128) void attn_kernel(
    const float* __restrict__ QKV, float* __restrict__ O)
{
    const int S = 1024, D = 1024, H3 = 3072, HD = 64;
    const int qb  = blockIdx.x;
    const int hh  = blockIdx.y;
    const int bsq = blockIdx.z;
    const int tid  = threadIdx.x;
    const int lane = tid & 31;
    const int wave = tid >> 5;
    const int half = (lane >> 4) & 1;
    const int l16  = lane & 15;

    __shared__ __attribute__((aligned(16))) unsigned short Qs[64][HD + 4];
    __shared__ __attribute__((aligned(16))) unsigned short Ks[64][HD + 4];
    __shared__ __attribute__((aligned(16))) unsigned short Vt[HD][64 + 4];   // [hd][kv]
    __shared__ __attribute__((aligned(16))) unsigned short Ps[4][16][HD + 4];

    const size_t base  = (size_t)bsq * S * H3;
    const float  scale = 0.125f;   // 1/sqrt(64), folded into Q

    for (int e = tid; e < 64 * HD; e += 128) {
        int r = e >> 6, c = e & 63;
        int t = qb * 64 + r;
        Qs[r][c] = f2bf(QKV[base + (size_t)t * H3 + hh * HD + c] * scale);
    }
    __syncthreads();

    Frag16 aq[2];
#pragma unroll
    for (int ks = 0; ks < 2; ++ks) {
        int m = wave * 16 + l16;
#pragma unroll
        for (int j = 0; j < 8; ++j)
            aq[ks].u[j] = *(const unsigned int*)&Qs[m][ks * 32 + kbase(j, half)];
    }

    const v8f zero8 = {0.f, 0.f, 0.f, 0.f, 0.f, 0.f, 0.f, 0.f};
    v8f oacc[4];
#pragma unroll
    for (int f = 0; f < 4; ++f) oacc[f] = zero8;
    float mstat[8], lstat[8];
#pragma unroll
    for (int i = 0; i < 8; ++i) { mstat[i] = -1e30f; lstat[i] = 0.f; }

    for (int kt = 0; kt <= qb; ++kt) {
        __syncthreads();
        // stage K tile [kv][hd] and V tile transposed [hd][kv]
        for (int e = tid; e < 64 * HD; e += 128) {
            int r = e >> 6, c = e & 63;
            int t = kt * 64 + r;
            size_t idx = base + (size_t)t * H3 + hh * HD + c;
            Ks[r][c] = f2bf(QKV[idx + 1024]);
            Vt[c][r] = f2bf(QKV[idx + 2048]);
        }
        __syncthreads();

        // ---- scores: s = (Q*scale) @ K^T  (16x64 per wave) ----
        v8f sacc[4];
#pragma unroll
        for (int f = 0; f < 4; ++f) sacc[f] = zero8;
#pragma unroll
        for (int f = 0; f < 4; ++f) {
#pragma unroll
            for (int ks = 0; ks < 2; ++ks) {
                Frag16 bk;
                int n = f * 16 + l16;  // kv column
#pragma unroll
                for (int j = 0; j < 8; ++j)
                    bk.u[j] = *(const unsigned int*)&Ks[n][ks * 32 + kbase(j, half)];
                sacc[f] = __builtin_amdgcn_wmma_f32_16x16x32_bf16(
                    false, aq[ks].v, false, bk.v, (short)0, sacc[f], false, false);
            }
        }

        // ---- mask (diagonal tile only) ----
        if (kt == qb) {
#pragma unroll
            for (int f = 0; f < 4; ++f)
#pragma unroll
                for (int i = 0; i < 8; ++i) {
                    int q = qb * 64 + wave * 16 + half * 8 + i;
                    int k = kt * 64 + f * 16 + l16;
                    bool ok = (k <= q) && !((q & 1) && (k == q - 1));
                    if (!ok) sacc[f][i] = -1e30f;
                }
        }

        // ---- online softmax (rows live in (vgpr, lane-half); reduce over 16 lanes) ----
        float rmax[8];
#pragma unroll
        for (int i = 0; i < 8; ++i)
            rmax[i] = fmaxf(fmaxf(sacc[0][i], sacc[1][i]), fmaxf(sacc[2][i], sacc[3][i]));
#pragma unroll
        for (int off = 8; off >= 1; off >>= 1)
#pragma unroll
            for (int i = 0; i < 8; ++i)
                rmax[i] = fmaxf(rmax[i], __shfl_xor(rmax[i], off, 32));

#pragma unroll
        for (int i = 0; i < 8; ++i) {
            float mnew = fmaxf(mstat[i], rmax[i]);
            float corr = expf(mstat[i] - mnew);
            mstat[i] = mnew;
            lstat[i] *= corr;
#pragma unroll
            for (int f = 0; f < 4; ++f) oacc[f][i] *= corr;
        }

        float rsum[8];
#pragma unroll
        for (int i = 0; i < 8; ++i) rsum[i] = 0.f;
#pragma unroll
        for (int f = 0; f < 4; ++f)
#pragma unroll
            for (int i = 0; i < 8; ++i) {
                float p = expf(sacc[f][i] - mstat[i]);
                rsum[i] += p;
                Ps[wave][half * 8 + i][f * 16 + l16] = f2bf(p);
            }
#pragma unroll
        for (int off = 8; off >= 1; off >>= 1)
#pragma unroll
            for (int i = 0; i < 8; ++i)
                rsum[i] += __shfl_xor(rsum[i], off, 32);
#pragma unroll
        for (int i = 0; i < 8; ++i) lstat[i] += rsum[i];

        // P was written in C-layout; wait for per-wave LDS stores to land
        asm volatile("s_wait_dscnt 0" ::: "memory");

        // ---- o += P @ V ----
#pragma unroll
        for (int ks = 0; ks < 2; ++ks) {
            Frag16 ap;
#pragma unroll
            for (int j = 0; j < 8; ++j)
                ap.u[j] = *(const unsigned int*)&Ps[wave][l16][ks * 32 + kbase(j, half)];
#pragma unroll
            for (int f = 0; f < 4; ++f) {
                Frag16 bv;
                int n = f * 16 + l16;  // hd column
#pragma unroll
                for (int j = 0; j < 8; ++j)
                    bv.u[j] = *(const unsigned int*)&Vt[n][ks * 32 + kbase(j, half)];
                oacc[f] = __builtin_amdgcn_wmma_f32_16x16x32_bf16(
                    false, ap.v, false, bv.v, (short)0, oacc[f], false, false);
            }
        }
    }

    // ---- write O[(b*S+q), h*HD+hd] = oacc / l ----
#pragma unroll
    for (int f = 0; f < 4; ++f)
#pragma unroll
        for (int i = 0; i < 8; ++i) {
            int q  = qb * 64 + wave * 16 + half * 8 + i;
            int hd = f * 16 + l16;
            O[((size_t)bsq * S + q) * D + hh * HD + hd] = oacc[f][i] / lstat[i];
        }
}

// ---------------------------------------------------------------------------
// LayerNorm: one block (256 threads) per row of 1024.
// ---------------------------------------------------------------------------
__global__ __launch_bounds__(256) void ln_kernel(
    const float* __restrict__ X, const float* __restrict__ g,
    const float* __restrict__ b, float* __restrict__ Y)
{
    const int Dd = 1024;
    int row = blockIdx.x;
    const float* x = X + (size_t)row * Dd;
    float* y = Y + (size_t)row * Dd;
    __shared__ float red[2][8];
    __shared__ float mv[2];
    float s = 0.f, s2 = 0.f;
    for (int i = threadIdx.x; i < Dd; i += 256) { float v = x[i]; s += v; s2 += v * v; }
#pragma unroll
    for (int off = 16; off >= 1; off >>= 1) {
        s  += __shfl_xor(s, off, 32);
        s2 += __shfl_xor(s2, off, 32);
    }
    int wv = threadIdx.x >> 5, ln = threadIdx.x & 31;
    if (ln == 0) { red[0][wv] = s; red[1][wv] = s2; }
    __syncthreads();
    if (threadIdx.x == 0) {
        float t = 0.f, t2 = 0.f;
        for (int w = 0; w < 8; ++w) { t += red[0][w]; t2 += red[1][w]; }
        float mu = t / (float)Dd;
        float var = t2 / (float)Dd - mu * mu;
        mv[0] = mu; mv[1] = rsqrtf(var + 1e-5f);
    }
    __syncthreads();
    float mu = mv[0], inv = mv[1];
    for (int i = threadIdx.x; i < Dd; i += 256)
        y[i] = (x[i] - mu) * inv * g[i] + b[i];
}

// ---------------------------------------------------------------------------
// Final LN + readout (D -> 1): one block per row.
// ---------------------------------------------------------------------------
__global__ __launch_bounds__(256) void final_kernel(
    const float* __restrict__ X, const float* __restrict__ g,
    const float* __restrict__ b, const float* __restrict__ wout,
    const float* __restrict__ bout, float* __restrict__ out)
{
    const int Dd = 1024;
    int row = blockIdx.x;
    const float* x = X + (size_t)row * Dd;
    __shared__ float red[3][8];
    __shared__ float mv[2];
    float s = 0.f, s2 = 0.f;
    for (int i = threadIdx.x; i < Dd; i += 256) { float v = x[i]; s += v; s2 += v * v; }
#pragma unroll
    for (int off = 16; off >= 1; off >>= 1) {
        s  += __shfl_xor(s, off, 32);
        s2 += __shfl_xor(s2, off, 32);
    }
    int wv = threadIdx.x >> 5, ln = threadIdx.x & 31;
    if (ln == 0) { red[0][wv] = s; red[1][wv] = s2; }
    __syncthreads();
    if (threadIdx.x == 0) {
        float t = 0.f, t2 = 0.f;
        for (int w = 0; w < 8; ++w) { t += red[0][w]; t2 += red[1][w]; }
        float mu = t / (float)Dd;
        float var = t2 / (float)Dd - mu * mu;
        mv[0] = mu; mv[1] = rsqrtf(var + 1e-5f);
    }
    __syncthreads();
    float mu = mv[0], inv = mv[1];
    float acc = 0.f;
    for (int i = threadIdx.x; i < Dd; i += 256)
        acc += ((x[i] - mu) * inv * g[i] + b[i]) * wout[i];
#pragma unroll
    for (int off = 16; off >= 1; off >>= 1) acc += __shfl_xor(acc, off, 32);
    if (ln == 0) red[2][wv] = acc;
    __syncthreads();
    if (threadIdx.x == 0) {
        float t = 0.f;
        for (int w = 0; w < 8; ++w) t += red[2][w];
        out[row] = t + bout[0];
    }
}

// ---------------------------------------------------------------------------
// Launcher
// ---------------------------------------------------------------------------
extern "C" void kernel_launch(void* const* d_in, const int* in_sizes, int n_in,
                              void* d_out, int out_size, void* d_ws, size_t ws_size,
                              hipStream_t stream) {
    (void)in_sizes; (void)n_in; (void)out_size; (void)ws_size;
    const int Mtok = 2048, Dd = 1024, L = 8, S = 1024;

    const float* x      = (const float*)d_in[0];
    // d_in[1] = attn_mask (unused; mask computed analytically)
    const float* win_w  = (const float*)d_in[2];
    const float* win_b  = (const float*)d_in[3];
    const float* wpe    = (const float*)d_in[4];
    const float* ln1_g  = (const float*)d_in[5];
    const float* ln1_b  = (const float*)d_in[6];
    const float* qkv_w  = (const float*)d_in[7];
    const float* qkv_b  = (const float*)d_in[8];
    const float* attn_w = (const float*)d_in[9];
    const float* attn_b = (const float*)d_in[10];
    const float* ln2_g  = (const float*)d_in[11];
    const float* ln2_b  = (const float*)d_in[12];
    const float* fc_w   = (const float*)d_in[13];
    const float* fc_b   = (const float*)d_in[14];
    const float* proj_w = (const float*)d_in[15];
    const float* proj_b = (const float*)d_in[16];
    const float* lnf_g  = (const float*)d_in[17];
    const float* lnf_b  = (const float*)d_in[18];
    const float* wout_w = (const float*)d_in[19];
    const float* wout_b = (const float*)d_in[20];

    float* h    = (float*)d_ws;                       // [2048,1024]
    float* a    = h    + (size_t)Mtok * Dd;           // [2048,1024]
    float* qkv  = a    + (size_t)Mtok * Dd;           // [2048,3072]
    float* attn = qkv  + (size_t)Mtok * 3 * Dd;       // [2048,1024]
    float* fc   = attn + (size_t)Mtok * Dd;           // [2048,4096]

    dim3 blk(256);

    // read-in: h = x @ win_w + win_b + wpe[pos]
    gemm_kernel<3><<<dim3(Dd / 128, Mtok / 128), blk, 0, stream>>>(
        x, win_w, win_b, wpe, h, Mtok, Dd, 64);

    for (int l = 0; l < L; ++l) {
        ln_kernel<<<Mtok, blk, 0, stream>>>(h, ln1_g + (size_t)l * Dd, ln1_b + (size_t)l * Dd, a);
        gemm_kernel<0><<<dim3(3 * Dd / 128, Mtok / 128), blk, 0, stream>>>(
            a, qkv_w + (size_t)l * Dd * 3 * Dd, qkv_b + (size_t)l * 3 * Dd, nullptr,
            qkv, Mtok, 3 * Dd, Dd);
        attn_kernel<<<dim3(S / 64, 16, 2), dim3(128), 0, stream>>>(qkv, attn);
        gemm_kernel<1><<<dim3(Dd / 128, Mtok / 128), blk, 0, stream>>>(
            attn, attn_w + (size_t)l * Dd * Dd, attn_b + (size_t)l * Dd, h,
            h, Mtok, Dd, Dd);
        ln_kernel<<<Mtok, blk, 0, stream>>>(h, ln2_g + (size_t)l * Dd, ln2_b + (size_t)l * Dd, a);
        gemm_kernel<2><<<dim3(4 * Dd / 128, Mtok / 128), blk, 0, stream>>>(
            a, fc_w + (size_t)l * Dd * 4 * Dd, fc_b + (size_t)l * 4 * Dd, nullptr,
            fc, Mtok, 4 * Dd, Dd);
        gemm_kernel<1><<<dim3(Dd / 128, Mtok / 128), blk, 0, stream>>>(
            fc, proj_w + (size_t)l * 4 * Dd * Dd, proj_b + (size_t)l * Dd, h,
            h, Mtok, Dd, 4 * Dd);
    }

    final_kernel<<<Mtok, blk, 0, stream>>>(h, lnf_g, lnf_b, wout_w, wout_b, (float*)d_out);
}
